// Attention_24395414241982
// MI455X (gfx1250) — matrix operand order repaired and checked by reading, not compiled, not run on hardware
//
#include <hip/hip_runtime.h>
#include <hip/hip_bf16.h>

#define ASYNC_STAGE 1

typedef __bf16 bf16;
typedef __attribute__((ext_vector_type(16))) __bf16 v16bf;
typedef __attribute__((ext_vector_type(8)))  __bf16 v8bf;
typedef __attribute__((ext_vector_type(4)))  __bf16 v4bf;
typedef __attribute__((ext_vector_type(8)))  float  v8f;
typedef __attribute__((ext_vector_type(4)))  float  v4f;

namespace {
constexpr int Bsz  = 32;
constexpr int Pn   = 577;
constexpr int Dm   = 768;
constexpr int Hh   = 12;
constexpr int Ss   = 64;
constexpr int Ppad = 608;              // ceil(577/32)*32
constexpr int Mtot = Bsz * Pn;         // 18464
constexpr int Mpad = 18496;            // 289 * 64
constexpr int N1   = 3 * Dm;           // 2304
constexpr float kScale = 0.10206207261596577f;  // (768/8)^-0.5 per reference bug

constexpr int LDS_STRIDE = 40;         // 32 + 8 pad (80B rows -> conflict-free frag reads)
constexpr int SA_BUF = 64  * LDS_STRIDE;   // elements per A buffer
constexpr int SB_BUF = 256 * LDS_STRIDE;   // elements per B buffer
}

union AFrag { v16bf v; v8bf h[2]; };

static __device__ inline v8f zero8() {
    v8f z = {0.f, 0.f, 0.f, 0.f, 0.f, 0.f, 0.f, 0.f};
    return z;
}

// A fragment (16x32 bf16) from row-major [M][ldk] global memory
static __device__ inline v16bf load_a_frag(const bf16* __restrict__ base, int ldk,
                                           int row0, int k0, int lane) {
    const int r  = row0 + (lane & 15);
    const int kk = k0 + ((lane >> 4) << 3);
    AFrag f;
    f.h[0] = *reinterpret_cast<const v8bf*>(base + (size_t)r * ldk + kk);
    f.h[1] = *reinterpret_cast<const v8bf*>(base + (size_t)r * ldk + kk + 16);
    return f.v;
}

// B fragment (32x16 bf16) from [N][ldk] (K contiguous) global memory
static __device__ inline v16bf load_b_frag(const bf16* __restrict__ base, int ldk,
                                           int col0, int k0, int lane) {
    const int n  = col0 + (lane & 15);
    const int kk = k0 + ((lane >> 4) << 3);
    AFrag f;
    f.h[0] = *reinterpret_cast<const v8bf*>(base + (size_t)n * ldk + kk);
    f.h[1] = *reinterpret_cast<const v8bf*>(base + (size_t)n * ldk + kk + 16);
    return f.v;
}

// Same fragment gathers but from an LDS tile with padded stride
static __device__ inline v16bf lds_a_frag(const bf16* sA, int row0, int lane) {
    const int r  = row0 + (lane & 15);
    const int kk = (lane >> 4) << 3;
    AFrag f;
    f.h[0] = *reinterpret_cast<const v8bf*>(sA + r * LDS_STRIDE + kk);
    f.h[1] = *reinterpret_cast<const v8bf*>(sA + r * LDS_STRIDE + kk + 16);
    return f.v;
}
static __device__ inline v16bf lds_b_frag(const bf16* sB, int col0, int lane) {
    const int n  = col0 + (lane & 15);
    const int kk = (lane >> 4) << 3;
    AFrag f;
    f.h[0] = *reinterpret_cast<const v8bf*>(sB + n * LDS_STRIDE + kk);
    f.h[1] = *reinterpret_cast<const v8bf*>(sB + n * LDS_STRIDE + kk + 16);
    return f.v;
}

static __device__ inline v8f wmma_bf16(v16bf a, v16bf b, v8f c) {
    return __builtin_amdgcn_wmma_f32_16x16x32_bf16(false, a, false, b,
                                                   (short)0, c, false, false);
}

#if ASYNC_STAGE
// Generic pointer to LDS: low 32 bits are the LDS byte offset (ISA §10.2 aperture rule)
static __device__ __forceinline__ unsigned lds_u32(const void* p) {
    return (unsigned)(unsigned long long)p;
}
static __device__ __forceinline__ void wait_async0() {
#if __has_builtin(__builtin_amdgcn_s_wait_asynccnt)
    __builtin_amdgcn_s_wait_asynccnt(0);
#else
    asm volatile("s_wait_asynccnt 0x0" ::: "memory");
#endif
}
// One k-step of cooperative staging: A = 16B/thread, B = 64B/thread.
// Instruction offset adds to BOTH global and LDS addresses (ISA §15.18 async pseudocode);
// B chunks are contiguous in both spaces so offset:0/16/32/48 is valid.
static __device__ __forceinline__ void async_tile_load(
    unsigned laA, unsigned laB, unsigned voA, unsigned voB,
    unsigned long long gbA, unsigned long long gbB)
{
    asm volatile("global_load_async_to_lds_b128 %0, %1, %2 offset:0"
                 :: "v"(laA), "v"(voA), "s"(gbA) : "memory");
    asm volatile("global_load_async_to_lds_b128 %0, %1, %2 offset:0"
                 :: "v"(laB), "v"(voB), "s"(gbB) : "memory");
    asm volatile("global_load_async_to_lds_b128 %0, %1, %2 offset:16"
                 :: "v"(laB), "v"(voB), "s"(gbB) : "memory");
    asm volatile("global_load_async_to_lds_b128 %0, %1, %2 offset:32"
                 :: "v"(laB), "v"(voB), "s"(gbB) : "memory");
    asm volatile("global_load_async_to_lds_b128 %0, %1, %2 offset:48"
                 :: "v"(laB), "v"(voB), "s"(gbB) : "memory");
}
#endif

// ---------------- shared LDS-tiled mainloop: block tile 64(M) x 256(N), K-step 32 ----------------
// 8 waves arranged 2(M) x 4(N); per-wave 32x64 accumulator (2x4 WMMA tiles).
// Double-buffered LDS; async copy overlaps WMMAs; last step peeled so the
// steady-state body is branch-free. compute_step keeps only 3 fragments live
// (a0, a1, one streamed B) to let the allocator keep accumulators in place.
static __device__ inline void gemm_mainloop_tiled(
    const bf16* __restrict__ Ag, const bf16* __restrict__ Bg,
    int gm0, int gn0, bf16* sA, bf16* sB,
    int tid, int lane, int wave, v8f acc[2][4])
{
    const int msub = wave >> 2;          // 0..1
    const int nsub = wave & 3;           // 0..3
    const int arow = tid >> 2;           // 0..63
    const int acol = (tid & 3) << 3;     // 0,8,16,24
    const int NK   = Dm / 32;            // 24

    auto compute_step = [&](const bf16* cA, const bf16* cB) {
        const v16bf a0 = lds_a_frag(cA, msub * 32,      lane);
        const v16bf a1 = lds_a_frag(cA, msub * 32 + 16, lane);
#pragma unroll
        for (int tn = 0; tn < 4; ++tn) {
            const v16bf bfr = lds_b_frag(cB, nsub * 64 + tn * 16, lane);
            acc[0][tn] = wmma_bf16(a0, bfr, acc[0][tn]);
            acc[1][tn] = wmma_bf16(a1, bfr, acc[1][tn]);
        }
    };

#if ASYNC_STAGE
    const unsigned long long gbA = (unsigned long long)(Ag + (size_t)gm0 * Dm);
    const unsigned long long gbB = (unsigned long long)(Bg + (size_t)gn0 * Dm);
    unsigned voA = (unsigned)((arow * Dm + acol) * 2);  // byte offsets vs SADDR base
    unsigned voB = (unsigned)(tid * Dm * 2);
    const unsigned laA0 = lds_u32(sA + arow * LDS_STRIDE + acol);
    const unsigned laA1 = laA0 + SA_BUF * 2;
    const unsigned laB0 = lds_u32(sB + tid * LDS_STRIDE);
    const unsigned laB1 = laB0 + SB_BUF * 2;

    async_tile_load(laA0, laB0, voA, voB, gbA, gbB);    // prefill buffer 0 (k-step 0)
    voA += 64; voB += 64;
    wait_async0();
    __syncthreads();

    for (int ks = 0; ks < NK - 1; ++ks) {
        const int nb = (ks + 1) & 1;
        async_tile_load(nb ? laA1 : laA0, nb ? laB1 : laB0, voA, voB, gbA, gbB);
        voA += 64; voB += 64;
        const int cb = ks & 1;
        compute_step(cb ? sA + SA_BUF : sA, cb ? sB + SB_BUF : sB);
        wait_async0();
        __syncthreads();
    }
    {
        const int cb = (NK - 1) & 1;
        compute_step(cb ? sA + SA_BUF : sA, cb ? sB + SB_BUF : sB);
    }
#else
    const bf16* gA = Ag + (size_t)(gm0 + arow) * Dm + acol;
    const bf16* gB = Bg + (size_t)(gn0 + tid) * Dm;
    bf16* sArow = sA + arow * LDS_STRIDE + acol;
    bf16* sBrow = sB + tid * LDS_STRIDE;
    {   // prefill buffer 0
        v8bf ra = *reinterpret_cast<const v8bf*>(gA);
        v8bf rb0 = *reinterpret_cast<const v8bf*>(gB);
        v8bf rb1 = *reinterpret_cast<const v8bf*>(gB + 8);
        v8bf rb2 = *reinterpret_cast<const v8bf*>(gB + 16);
        v8bf rb3 = *reinterpret_cast<const v8bf*>(gB + 24);
        *reinterpret_cast<v8bf*>(sArow) = ra;
        *reinterpret_cast<v8bf*>(sBrow)      = rb0;
        *reinterpret_cast<v8bf*>(sBrow + 8)  = rb1;
        *reinterpret_cast<v8bf*>(sBrow + 16) = rb2;
        *reinterpret_cast<v8bf*>(sBrow + 24) = rb3;
    }
    __syncthreads();
    for (int ks = 0; ks < NK - 1; ++ks) {
        const int k0 = (ks + 1) * 32;
        v8bf ra  = *reinterpret_cast<const v8bf*>(gA + k0);
        v8bf rb0 = *reinterpret_cast<const v8bf*>(gB + k0);
        v8bf rb1 = *reinterpret_cast<const v8bf*>(gB + k0 + 8);
        v8bf rb2 = *reinterpret_cast<const v8bf*>(gB + k0 + 16);
        v8bf rb3 = *reinterpret_cast<const v8bf*>(gB + k0 + 24);
        const int cb = ks & 1;
        compute_step(cb ? sA + SA_BUF : sA, cb ? sB + SB_BUF : sB);
        const int nx = (ks + 1) & 1;
        bf16* dA = sArow + nx * SA_BUF;
        bf16* dB = sBrow + nx * SB_BUF;
        *reinterpret_cast<v8bf*>(dA) = ra;
        *reinterpret_cast<v8bf*>(dB)      = rb0;
        *reinterpret_cast<v8bf*>(dB + 8)  = rb1;
        *reinterpret_cast<v8bf*>(dB + 16) = rb2;
        *reinterpret_cast<v8bf*>(dB + 24) = rb3;
        __syncthreads();
    }
    {
        const int cb = (NK - 1) & 1;
        compute_step(cb ? sA + SA_BUF : sA, cb ? sB + SB_BUF : sB);
    }
#endif
}

// ---------------- f32 -> bf16 convert (vectorized x4) ----------------
__global__ void cvt_f32_bf16_x4(const float* __restrict__ src, bf16* __restrict__ dst, int n4) {
    int i = blockIdx.x * blockDim.x + threadIdx.x;
    const int stride = gridDim.x * blockDim.x;
    for (; i < n4; i += stride) {
        v4f f = reinterpret_cast<const v4f*>(src)[i];
        v4bf o;
        o.x = (bf16)f.x; o.y = (bf16)f.y; o.z = (bf16)f.z; o.w = (bf16)f.w;
        reinterpret_cast<v4bf*>(dst)[i] = o;
    }
}

// Zero padding columns (p in [577,608)) of vT[b,h,s,p]
__global__ void zero_vpad(bf16* __restrict__ vb) {
    const int pad   = Ppad - Pn;                // 31
    const int rows  = Bsz * Hh * Ss;            // 24576
    const int total = rows * pad;
    int i = blockIdx.x * blockDim.x + threadIdx.x;
    if (i < total) {
        const int r = i / pad, c = i - r * pad;
        vb[(size_t)r * Ppad + Pn + c] = (bf16)0.0f;
    }
}

// ---------------- GEMM1: qkv = x @ qkv_w^T + b ; scatter to q/k/vT ----------------
__global__ void __launch_bounds__(256) gemm_qkv(
    const bf16* __restrict__ xb, const bf16* __restrict__ wq,
    const float* __restrict__ qkv_b,
    bf16* __restrict__ qb, bf16* __restrict__ kb, bf16* __restrict__ vb)
{
    __shared__ __align__(16) bf16 sA[2 * SA_BUF];
    __shared__ __align__(16) bf16 sB[2 * SB_BUF];
    const int tid  = threadIdx.x;
    const int lane = tid & 31;
    const int wave = tid >> 5;
    const int NB = N1 / 256;                       // 9
    const int mb = blockIdx.x / NB, nb = blockIdx.x % NB;
    const int gm0 = mb * 64, gn0 = nb * 256;

    v8f acc[2][4];
#pragma unroll
    for (int a = 0; a < 2; ++a)
#pragma unroll
        for (int b = 0; b < 4; ++b) acc[a][b] = zero8();

    gemm_mainloop_tiled(xb, wq, gm0, gn0, sA, sB, tid, lane, wave, acc);

    const int n0w = gn0 + (wave & 3) * 64;         // 64-aligned => (c,h) constant
    const int m0w = gm0 + (wave >> 2) * 32;
    const int c   = n0w / Dm;                      // 0=v, 1=q, 2=k (reference split order)
    const int hh  = (n0w % Dm) / Ss;
#pragma unroll
    for (int tm = 0; tm < 2; ++tm) {
#pragma unroll
        for (int tn = 0; tn < 4; ++tn) {
            const int n = n0w + tn * 16 + (lane & 15);
            const int s = tn * 16 + (lane & 15);
            const float bias = qkv_b[n];
#pragma unroll
            for (int r = 0; r < 8; ++r) {
                const int m = m0w + tm * 16 + r + ((lane >> 4) << 3);
                if (m < Mtot) {
                    const int b = m / Pn;
                    const int p = m - b * Pn;
                    const bf16 bv = (bf16)(acc[tm][tn][r] + bias);
                    const int bh = b * Hh + hh;
                    if (c == 1)      qb[((size_t)bh * Ppad + p) * Ss + s] = bv;
                    else if (c == 2) kb[((size_t)bh * Ppad + p) * Ss + s] = bv;
                    else             vb[((size_t)bh * Ss + s) * Ppad + p] = bv;
                }
            }
        }
    }
}

// ---------------- Fused attention: per (b,h, 16-row block) ----------------
__global__ void __launch_bounds__(256) attn_fused(
    const bf16* __restrict__ qb, const bf16* __restrict__ kb,
    const bf16* __restrict__ vb, bf16* __restrict__ ob)
{
    __shared__ __align__(16) float sc[16][Ppad];   // 38 KB (reused as reduction scratch)
    __shared__ __align__(16) bf16  pr[16][Ppad];   // 19 KB

    const int lane = threadIdx.x & 31;
    const int wave = threadIdx.x >> 5;
    const int nPB  = (Pn + 15) / 16;               // 37
    const int pblk = blockIdx.x % nPB;
    const int bh   = blockIdx.x / nPB;             // 0..383
    const int h    = bh % Hh;
    const int b    = bh / Hh;
    const int p0   = pblk * 16;

    const bf16* qrow = qb + (size_t)bh * Ppad * Ss;
    const bf16* krow = kb + (size_t)bh * Ppad * Ss;
    const bf16* vrow = vb + (size_t)bh * Ss * Ppad;

    // ---- phase 1: scores = Q @ K^T  (16 x 608) ----
    const v16bf qa0 = load_a_frag(qrow, Ss, p0, 0,  lane);
    const v16bf qa1 = load_a_frag(qrow, Ss, p0, 32, lane);
    for (int ci = wave; ci < Ppad / 16; ci += 8) {
        v8f accs = zero8();
        const v16bf b0 = load_b_frag(krow, Ss, ci * 16, 0,  lane);
        const v16bf b1 = load_b_frag(krow, Ss, ci * 16, 32, lane);
        accs = wmma_bf16(qa0, b0, accs);
        accs = wmma_bf16(qa1, b1, accs);
#pragma unroll
        for (int r = 0; r < 8; ++r) {
            const int row = r + ((lane >> 4) << 3);
            const int col = ci * 16 + (lane & 15);
            sc[row][col] = accs[r];
        }
    }
    __syncthreads();

    // ---- phase 2: masked softmax over 577 cols (16 threads per row) ----
    {
        const int row = threadIdx.x >> 4;   // 0..15
        const int tc  = threadIdx.x & 15;
        float mx = -3.0e38f;
        for (int i = 0; i < Ppad / 16; ++i) {
            const int cc = tc + 16 * i;
            if (cc < Pn) mx = fmaxf(mx, sc[row][cc] * kScale);
        }
#pragma unroll
        for (int off = 8; off >= 1; off >>= 1) mx = fmaxf(mx, __shfl_xor(mx, off, 16));
        float sum = 0.f;
        for (int i = 0; i < Ppad / 16; ++i) {
            const int cc = tc + 16 * i;
            if (cc < Pn) sum += __expf(sc[row][cc] * kScale - mx);
        }
#pragma unroll
        for (int off = 8; off >= 1; off >>= 1) sum += __shfl_xor(sum, off, 16);
        const float inv = 1.0f / sum;
        for (int i = 0; i < Ppad / 16; ++i) {
            const int cc = tc + 16 * i;
            const float pv = (cc < Pn) ? __expf(sc[row][cc] * kScale - mx) * inv : 0.f;
            pr[row][cc] = (bf16)pv;
        }
    }
    __syncthreads();

    // ---- phase 3: out = P @ V over all 8 waves (K split in halves) ----
    {
        const int ntile = wave & 3;                 // s-tile
        const int half  = wave >> 2;                // K half
        const int n0    = ntile * 16;
        const int ksBeg = half ? 10 : 0;
        const int ksEnd = half ? 19 : 10;           // 19 k-steps total (608/32)
        v8f acco = zero8();
        for (int ks = ksBeg; ks < ksEnd; ++ks) {
            const int ar = lane & 15;
            const int kk = ks * 32 + ((lane >> 4) << 3);
            AFrag af;
            af.h[0] = *reinterpret_cast<const v8bf*>(&pr[ar][kk]);
            af.h[1] = *reinterpret_cast<const v8bf*>(&pr[ar][kk + 16]);
            const v16bf bf_ = load_b_frag(vrow, Ppad, n0, ks * 32, lane);
            acco = wmma_bf16(af.v, bf_, acco);
        }
        float* red = &sc[0][0];                     // sc is dead now
        if (half) {
#pragma unroll
            for (int r = 0; r < 8; ++r) {
                const int row = r + ((lane >> 4) << 3);
                red[(ntile * 16 + row) * 16 + (lane & 15)] = acco[r];
            }
        }
        __syncthreads();
        if (!half) {
#pragma unroll
            for (int r = 0; r < 8; ++r) {
                const int row = r + ((lane >> 4) << 3);
                acco[r] += red[(ntile * 16 + row) * 16 + (lane & 15)];
                const int p = p0 + row;
                if (p < Pn) {
                    const int d = h * Ss + n0 + (lane & 15);
                    ob[((size_t)(b * Pn + p)) * Dm + d] = (bf16)acco[r];
                }
            }
        }
    }
}

// ---------------- GEMM2: out = attn_out @ out_w^T + out_b (f32 store) ----------------
__global__ void __launch_bounds__(256) gemm_out(
    const bf16* __restrict__ ab, const bf16* __restrict__ wo,
    const float* __restrict__ bias, float* __restrict__ out)
{
    __shared__ __align__(16) bf16 sA[2 * SA_BUF];
    __shared__ __align__(16) bf16 sB[2 * SB_BUF];
    const int tid  = threadIdx.x;
    const int lane = tid & 31;
    const int wave = tid >> 5;
    const int NB = Dm / 256;                       // 3
    const int mb = blockIdx.x / NB, nb = blockIdx.x % NB;
    const int gm0 = mb * 64, gn0 = nb * 256;

    v8f acc[2][4];
#pragma unroll
    for (int a = 0; a < 2; ++a)
#pragma unroll
        for (int b = 0; b < 4; ++b) acc[a][b] = zero8();

    gemm_mainloop_tiled(ab, wo, gm0, gn0, sA, sB, tid, lane, wave, acc);

    const int n0w = gn0 + (wave & 3) * 64;
    const int m0w = gm0 + (wave >> 2) * 32;
#pragma unroll
    for (int tm = 0; tm < 2; ++tm) {
#pragma unroll
        for (int tn = 0; tn < 4; ++tn) {
            const int n = n0w + tn * 16 + (lane & 15);
            const float bv = bias[n];
#pragma unroll
            for (int r = 0; r < 8; ++r) {
                const int m = m0w + tm * 16 + r + ((lane >> 4) << 3);
                if (m < Mtot) out[(size_t)m * Dm + n] = acc[tm][tn][r] + bv;
            }
        }
    }
}

extern "C" void kernel_launch(void* const* d_in, const int* in_sizes, int n_in,
                              void* d_out, int out_size, void* d_ws, size_t ws_size,
                              hipStream_t stream) {
    (void)in_sizes; (void)n_in; (void)out_size; (void)ws_size;
    const float* x     = (const float*)d_in[0];
    const float* qkv_w = (const float*)d_in[1];
    const float* qkv_b = (const float*)d_in[2];
    const float* out_w = (const float*)d_in[3];
    const float* out_b = (const float*)d_in[4];
    float* out = (float*)d_out;

    char* ws = (char*)d_ws;
    size_t off = 0;
    auto carve = [&](size_t bytes) -> void* {
        void* p = ws + off;
        off = (off + bytes + 255) & ~(size_t)255;
        return p;
    };
    bf16* xb = (bf16*)carve((size_t)Mpad * Dm * 2);   // padded; reused as attention output
    bf16* wq = (bf16*)carve((size_t)N1 * Dm * 2);
    bf16* wo = (bf16*)carve((size_t)Dm * Dm * 2);
    const size_t qkvBytes = (size_t)Bsz * Hh * Ppad * Ss * 2;
    bf16* qb = (bf16*)carve(qkvBytes);
    bf16* kb = (bf16*)carve(qkvBytes);
    bf16* vb = (bf16*)carve(qkvBytes);

    // 1) converts
    {
        const int n4x = Mtot * Dm / 4;
        cvt_f32_bf16_x4<<<(n4x + 255) / 256, 256, 0, stream>>>(x, xb, n4x);
        const int n4q = N1 * Dm / 4;
        cvt_f32_bf16_x4<<<(n4q + 255) / 256, 256, 0, stream>>>(qkv_w, wq, n4q);
        const int n4o = Dm * Dm / 4;
        cvt_f32_bf16_x4<<<(n4o + 255) / 256, 256, 0, stream>>>(out_w, wo, n4o);
    }

    // 2) zero vT pad columns (avoid 0 * NaN in P@V)
    {
        const int total = Bsz * Hh * Ss * (Ppad - Pn);
        zero_vpad<<<(total + 255) / 256, 256, 0, stream>>>(vb);
    }

    // 3) QKV projection (LDS-tiled, async-staged)
    {
        const int blocks = (Mpad / 64) * (N1 / 256);  // 289 * 9 = 2601
        gemm_qkv<<<blocks, 256, 0, stream>>>(xb, wq, qkv_b, qb, kb, vb);
    }

    // 4) fused attention (writes bf16 attention output into xb)
    {
        const int blocks = Bsz * Hh * ((Pn + 15) / 16);  // 14208
        attn_fused<<<blocks, 256, 0, stream>>>(qb, kb, vb, xb);
    }

    // 5) output projection (LDS-tiled, async-staged)
    {
        const int blocks = (Mpad / 64) * (Dm / 256);  // 289 * 3 = 867
        gemm_out<<<blocks, 256, 0, stream>>>(xb, wo, out_b, out);
    }
}
